// MoETransformerBlock_46084999086875
// MI455X (gfx1250) — matrix operand order repaired and checked by reading, not compiled, not run on hardware
//
#include <hip/hip_runtime.h>
#include <hip/hip_bf16.h>
#include <math.h>

// ---------------- problem constants ----------------
#define B_   4
#define S_   2048
#define D_   1024
#define H_   16
#define KV_  4
#define HD_  64
#define E_   8
#define F_   1024
#define NTOK (B_*S_)
#define CAP  4096          // per-expert token capacity (2x expected load)
#define EPS  1e-6f

// ---------------- GEMM tiling ----------------
#define BM 128
#define BN 128
#define BK 32
#define LDT 40             // LDS tile row stride in elements (BK + 8 pad)

typedef __attribute__((ext_vector_type(16))) __bf16 v16bf;
typedef __attribute__((ext_vector_type(8)))  float  v8f;

static __device__ __forceinline__ float  bf2f(__bf16 x){ return (float)x; }
static __device__ __forceinline__ __bf16 f2bf(float  x){ return (__bf16)x; }

static __device__ __forceinline__ v8f wmma_bf16(v16bf a, v16bf b, v8f c){
  // D(f32 16x16) = A(bf16 16x32) * B(bf16 32x16) + C
  return __builtin_amdgcn_wmma_f32_16x16x32_bf16(false, a, false, b, (short)0, c,
                                                 false, false);
}

// ---------------- CDNA5 async global->LDS copy (ASYNCcnt path) ----------------
// Copies 32 bytes per lane: lane-private global addr -> lane-private LDS addr.
// LDS byte address = low 32 bits of the generic pointer (ISA: LDS aperture
// addresses truncate to addr[31:0]).
static __device__ __forceinline__ void async_copy32(void* lds, const void* g){
  unsigned      l = (unsigned)(unsigned long long)(uintptr_t)lds;
  unsigned long long a = (unsigned long long)(uintptr_t)g;
  asm volatile("global_load_async_to_lds_b128 %0, %1, off\n\t"
               "global_load_async_to_lds_b128 %0, %1, off offset:16"
               :: "v"(l), "v"(a) : "memory");
}
static __device__ __forceinline__ void async_wait(){
  asm volatile("s_wait_asynccnt 0x0" ::: "memory");
}

// A-matrix 16-bit fragment: lane holds K {k0..k0+15} (contiguous),
// k0 = (lane>>4)*16. Caller passes p = &row[k0].
static __device__ __forceinline__ v16bf fragA(const __bf16* p){
  v16bf f;
#pragma unroll
  for (int i = 0; i < 8; ++i) f[i]     = p[i];
#pragma unroll
  for (int i = 0; i < 8; ++i) f[8 + i] = p[8 + i];
  return f;
}

// B-matrix 16-bit fragment: lanes 0-15 hold K=0..15, lanes 16-31 hold K=16..31
// (contiguous per lane). Caller passes p = &col_row[(lane>>4)*16].
static __device__ __forceinline__ v16bf fragB(const __bf16* p){
  v16bf f;
#pragma unroll
  for (int i = 0; i < 16; ++i) f[i] = p[i];
  return f;
}

// =====================================================================
// Flexible WMMA GEMM: C(128x128) per block, 8 waves of 4x2 WMMA tiles.
//   mode 0: out_bf16 = A @ W                         (Q, K projections)
//   mode 1: V projection -> f32 v output + bf16 V^T
//   mode 2: h = resid + A @ Wo  -> f32 (ws h and d_out)
//   mode 3: MoE stage1: he = silu(A@Wg) * (A@Wu), A rows gathered by list
//   mode 4: MoE stage2: atomic scatter out += gate * (he @ Wd)
// =====================================================================
struct GemmP {
  int mode;
  const __bf16* A;  long lda;  long a_estride;
  const float*  W;  long w_estride; int ldw;
  const float*  W2;
  int K;
  __bf16* outB; int ldo;
  const float* resid; float* outF; float* outF2;
  const int* rowlist; const float* glist; const int* cnts;
  float* scat;
  float* vF; __bf16* vT;
};

__global__ void __launch_bounds__(256) gemm_kernel(GemmP p){
  __shared__ __bf16 As [BM * LDT];
  __shared__ __bf16 Bs [BN * LDT];
  __shared__ __bf16 B2s[BN * LDT];

  const int e = blockIdx.z;
  int count = 0;
  if (p.mode >= 3){
    count = p.cnts[e];
    if (count > CAP) count = CAP;
    if ((int)blockIdx.x * BM >= count) return;   // uniform per block
  }
  const int rowBase = blockIdx.x * BM;
  const int colBase = blockIdx.y * BN;
  const float*  W  = p.W + (long)e * p.w_estride;
  const float*  W2 = p.W2 ? (p.W2 + (long)e * p.w_estride) : (const float*)0;
  const __bf16* A  = p.A + (long)e * p.a_estride;

  const int tid  = threadIdx.x;
  const int lane = tid & 31;
  const int w    = tid >> 5;
  const int wm   = w >> 2;      // 0..1 -> 64-row half
  const int wn   = w & 3;       // 0..3 -> 32-col quarter

  v8f acc[4][2], acc2[4][2];
#pragma unroll
  for (int i = 0; i < 4; ++i)
#pragma unroll
    for (int j = 0; j < 2; ++j)
#pragma unroll
      for (int r = 0; r < 8; ++r){ acc[i][j][r] = 0.f; acc2[i][j][r] = 0.f; }

  // A staging: thread -> (row, 16-elem segment)
  const int arow = tid >> 1;
  const int aseg = (tid & 1) * 16;
  long asrc;
  if (p.mode == 3){
    int pos = rowBase + arow;
    int pc  = (count > 0) ? (pos < count ? pos : count - 1) : 0;
    asrc = (long)p.rowlist[e * CAP + pc] * p.lda;
  } else {
    asrc = (long)(rowBase + arow) * p.lda;
  }
  // B staging: thread -> (k row, 16-col segment), transpose+cvt into LDS
  const int bk  = tid >> 3;         // 0..31
  const int bn0 = (tid & 7) * 16;   // 0..112

  for (int k0 = 0; k0 < p.K; k0 += BK){
    // A tile (bf16, contiguous): async global->LDS, 32B per lane
    async_copy32(&As[arow * LDT + aseg], A + asrc + k0 + aseg);
    if (k0 + BK < p.K) __builtin_prefetch(A + asrc + k0 + BK + aseg, 0, 0);
    { // B tile: f32 load -> bf16, transposed [n][k]
      const float* gw = W + (long)(k0 + bk) * p.ldw + colBase + bn0;
#pragma unroll
      for (int i = 0; i < 16; ++i) Bs[(bn0 + i) * LDT + bk] = f2bf(gw[i]);
      if (p.mode == 3){
        const float* gw2 = W2 + (long)(k0 + bk) * p.ldw + colBase + bn0;
#pragma unroll
        for (int i = 0; i < 16; ++i) B2s[(bn0 + i) * LDT + bk] = f2bf(gw2[i]);
      }
      if (k0 + BK < p.K) __builtin_prefetch(gw + (long)BK * p.ldw, 0, 0);
    }
    async_wait();
    __syncthreads();

    const int kl = (lane >> 4) * 16;   // A-frag K base
    const int kb = (lane >> 4) * 16;   // B-frag K base
    v16bf af[4];
#pragma unroll
    for (int i = 0; i < 4; ++i)
      af[i] = fragA(&As[(wm * 64 + i * 16 + (lane & 15)) * LDT + kl]);
#pragma unroll
    for (int j = 0; j < 2; ++j){
      v16bf bf = fragB(&Bs[(wn * 32 + j * 16 + (lane & 15)) * LDT + kb]);
#pragma unroll
      for (int i = 0; i < 4; ++i) acc[i][j] = wmma_bf16(af[i], bf, acc[i][j]);
      if (p.mode == 3){
        v16bf bf2 = fragB(&B2s[(wn * 32 + j * 16 + (lane & 15)) * LDT + kb]);
#pragma unroll
        for (int i = 0; i < 4; ++i) acc2[i][j] = wmma_bf16(af[i], bf2, acc2[i][j]);
      }
    }
    __syncthreads();
  }

  // ---------------- epilogue ----------------
#pragma unroll
  for (int i = 0; i < 4; ++i){
    const int mbase = wm * 64 + i * 16 + (lane >> 4) * 8;
#pragma unroll
    for (int j = 0; j < 2; ++j){
      const int col = colBase + wn * 32 + j * 16 + (lane & 15);
#pragma unroll
      for (int r = 0; r < 8; ++r){
        const int row = rowBase + mbase + r;
        const float v = acc[i][j][r];
        if (p.mode == 0){
          p.outB[(long)row * p.ldo + col] = f2bf(v);
        } else if (p.mode == 1){
          p.vF[(long)row * (KV_ * HD_) + col] = v;           // f32 v out
          const int bb = row >> 11, ss = row & (S_ - 1);
          const int kv = col >> 6,  hd = col & 63;
          p.vT[(((long)(bb * KV_ + kv) * HD_ + hd) * S_) + ss] = f2bf(v);
        } else if (p.mode == 2){
          const float hv = p.resid[(long)row * D_ + col] + v;
          p.outF [(long)row * D_ + col] = hv;                // h (ws)
          p.outF2[(long)row * D_ + col] = hv;                // d_out (moe adds)
        } else if (p.mode == 3){
          const float g  = v;
          const float u  = acc2[i][j][r];
          const float he = (g / (1.0f + __expf(-g))) * u;    // silu(g)*u
          p.outB[((long)e * CAP + row) * p.ldo + col] = f2bf(he);
        } else { // mode 4
          if (row < count){
            const int   token = p.rowlist[e * CAP + row];
            const float gate  = p.glist  [e * CAP + row];
            atomicAdd(&p.scat[(long)token * D_ + col], gate * v);
          }
        }
      }
    }
  }
}

// =====================================================================
// RMSNorm (f32 in -> bf16 out), one block per token row
// =====================================================================
__global__ void __launch_bounds__(256) rmsnorm_kernel(const float* __restrict__ x,
                                                      const float* __restrict__ w,
                                                      __bf16* __restrict__ out){
  __shared__ float red[8];
  const long row = blockIdx.x;
  const float4 xv = reinterpret_cast<const float4*>(x + row * D_)[threadIdx.x];
  float ss = xv.x*xv.x + xv.y*xv.y + xv.z*xv.z + xv.w*xv.w;
#pragma unroll
  for (int m = 16; m >= 1; m >>= 1) ss += __shfl_xor(ss, m, 32);
  if ((threadIdx.x & 31) == 0) red[threadIdx.x >> 5] = ss;
  __syncthreads();
  float tot = 0.f;
#pragma unroll
  for (int i = 0; i < 8; ++i) tot += red[i];
  const float sc = rsqrtf(tot / (float)D_ + EPS);
  const float4 wv = reinterpret_cast<const float4*>(w)[threadIdx.x];
  const int c = threadIdx.x * 4;
  out[row * D_ + c + 0] = f2bf(xv.x * sc * wv.x);
  out[row * D_ + c + 1] = f2bf(xv.y * sc * wv.y);
  out[row * D_ + c + 2] = f2bf(xv.z * sc * wv.z);
  out[row * D_ + c + 3] = f2bf(xv.w * sc * wv.w);
}

// =====================================================================
// RMSNorm + router + top-2 softmax + per-expert list build
// =====================================================================
__global__ void __launch_bounds__(256) rms_router_kernel(const float* __restrict__ h,
                                                         const float* __restrict__ w,
                                                         const float* __restrict__ rw,
                                                         __bf16* __restrict__ y,
                                                         int* cnts, int* list, float* glist){
  __shared__ float red[8];
  __shared__ float lred[8][8];
  const int token = blockIdx.x;
  const float4 xv = reinterpret_cast<const float4*>(h + (long)token * D_)[threadIdx.x];
  float ss = xv.x*xv.x + xv.y*xv.y + xv.z*xv.z + xv.w*xv.w;
#pragma unroll
  for (int m = 16; m >= 1; m >>= 1) ss += __shfl_xor(ss, m, 32);
  if ((threadIdx.x & 31) == 0) red[threadIdx.x >> 5] = ss;
  __syncthreads();
  float tot = 0.f;
#pragma unroll
  for (int i = 0; i < 8; ++i) tot += red[i];
  const float sc = rsqrtf(tot / (float)D_ + EPS);
  const float4 wv = reinterpret_cast<const float4*>(w)[threadIdx.x];
  const int c = threadIdx.x * 4;
  float yv[4] = { xv.x*sc*wv.x, xv.y*sc*wv.y, xv.z*sc*wv.z, xv.w*sc*wv.w };
#pragma unroll
  for (int i = 0; i < 4; ++i) y[(long)token * D_ + c + i] = f2bf(yv[i]);

  float le[8];
#pragma unroll
  for (int e = 0; e < 8; ++e) le[e] = 0.f;
#pragma unroll
  for (int i = 0; i < 4; ++i){
    const float4* rp = reinterpret_cast<const float4*>(rw + (long)(c + i) * E_);
    const float4 r0 = rp[0], r1 = rp[1];
    le[0] += yv[i]*r0.x; le[1] += yv[i]*r0.y; le[2] += yv[i]*r0.z; le[3] += yv[i]*r0.w;
    le[4] += yv[i]*r1.x; le[5] += yv[i]*r1.y; le[6] += yv[i]*r1.z; le[7] += yv[i]*r1.w;
  }
#pragma unroll
  for (int e = 0; e < 8; ++e)
#pragma unroll
    for (int m = 16; m >= 1; m >>= 1) le[e] += __shfl_xor(le[e], m, 32);
  if ((threadIdx.x & 31) == 0)
#pragma unroll
    for (int e = 0; e < 8; ++e) lred[threadIdx.x >> 5][e] = le[e];
  __syncthreads();
  if (threadIdx.x == 0){
    float lg[8];
#pragma unroll
    for (int e = 0; e < 8; ++e){
      float s = 0.f;
#pragma unroll
      for (int q = 0; q < 8; ++q) s += lred[q][e];
      lg[e] = s;
    }
    int i0 = 0;
#pragma unroll
    for (int e = 1; e < 8; ++e) if (lg[e] > lg[i0]) i0 = e;
    int i1 = (i0 == 0) ? 1 : 0;
#pragma unroll
    for (int e = 0; e < 8; ++e) if (e != i0 && lg[e] > lg[i1]) i1 = e;
    const float e1 = __expf(lg[i1] - lg[i0]);
    const float inv = 1.0f / (1.0f + e1);
    const float g0 = inv, g1 = e1 * inv;
    int p0 = atomicAdd(&cnts[i0], 1);
    if (p0 < CAP){ list[i0 * CAP + p0] = token; glist[i0 * CAP + p0] = g0; }
    int p1 = atomicAdd(&cnts[i1], 1);
    if (p1 < CAP){ list[i1 * CAP + p1] = token; glist[i1 * CAP + p1] = g1; }
  }
}

// =====================================================================
// RoPE kernels
// =====================================================================
__global__ void __launch_bounds__(256) rope_q_kernel(__bf16* q, const float* cb, const float* sb){
  const int t = blockIdx.x * 256 + threadIdx.x;     // NTOK*H_*32 threads
  const int token = t >> 9;
  const int rem = t & 511;
  const int hh = rem >> 5, hd = rem & 31;
  const int s = token & (S_ - 1);
  const long base = (long)token * (H_ * HD_) + hh * HD_ + hd;
  const float q1 = bf2f(q[base]), q2 = bf2f(q[base + 32]);
  const float c1 = cb[s * HD_ + hd], c2 = cb[s * HD_ + hd + 32];
  const float s1 = sb[s * HD_ + hd], s2 = sb[s * HD_ + hd + 32];
  q[base]      = f2bf(q1 * c1 - q2 * s1);
  q[base + 32] = f2bf(q2 * c2 + q1 * s2);
}

__global__ void __launch_bounds__(256) rope_k_kernel(__bf16* k, const float* cb, const float* sb,
                                                     float* kout){
  const int t = blockIdx.x * 256 + threadIdx.x;     // NTOK*KV_*32 threads
  const int token = t >> 7;
  const int rem = t & 127;
  const int kv = rem >> 5, hd = rem & 31;
  const int s = token & (S_ - 1);
  const long base = (long)token * (KV_ * HD_) + kv * HD_ + hd;
  const float k1 = bf2f(k[base]), k2 = bf2f(k[base + 32]);
  const float c1 = cb[s * HD_ + hd], c2 = cb[s * HD_ + hd + 32];
  const float s1 = sb[s * HD_ + hd], s2 = sb[s * HD_ + hd + 32];
  const float o1 = k1 * c1 - k2 * s1;
  const float o2 = k2 * c2 + k1 * s2;
  k[base]      = f2bf(o1);
  k[base + 32] = f2bf(o2);
  kout[base]      = o1;       // f32 k output (post-RoPE, per reference)
  kout[base + 32] = o2;
}

// =====================================================================
// Flash attention (causal GQA): block = (128 q rows, head, batch),
// 8 waves x 16 q rows each; K/V^T tiles async-staged in LDS; WMMA everywhere.
// =====================================================================
__global__ void __launch_bounds__(256) attn_kernel(const __bf16* __restrict__ q,
                                                   const __bf16* __restrict__ k,
                                                   const __bf16* __restrict__ vT,
                                                   __bf16* __restrict__ o){
  __shared__ __bf16 Ks[64 * 72];          // [key][hd]
  __shared__ __bf16 Vs[64 * 72];          // [hd][key]
  __shared__ __bf16 Ps[8 * 16 * 72];      // per-wave P tile [row][key]
  const int b = blockIdx.z, hh = blockIdx.y, qt = blockIdx.x;
  const int kvh = hh >> 2;                // H/KV = 4
  const int tid = threadIdx.x, lane = tid & 31, w = tid >> 5;
  const int qBase = qt * 128 + w * 16;
  const int mb = (lane >> 4) * 8;         // C-layout row base for this lane
  const int kl = (lane >> 4) * 16;        // A-frag K base
  const int kb = (lane >> 4) * 16;        // B-frag K base

  // q A-fragments held in registers for the whole loop
  const __bf16* qrow = q + ((long)(b * S_ + qBase + (lane & 15))) * (H_ * HD_) + hh * HD_;
  v16bf qf[2];
  qf[0] = fragA(qrow + kl);
  qf[1] = fragA(qrow + 32 + kl);

  float mrow[8], lrow[8];
  v8f oa[4];
#pragma unroll
  for (int r = 0; r < 8; ++r){ mrow[r] = -1e30f; lrow[r] = 0.f; }
#pragma unroll
  for (int t4 = 0; t4 < 4; ++t4)
#pragma unroll
    for (int r = 0; r < 8; ++r) oa[t4][r] = 0.f;

  const int kEnd = qt * 128 + 128;        // causal upper bound for this block
  for (int kt = 0; kt < kEnd; kt += 64){
    { // async stage K tile [64 keys][64 hd] and V^T tile [64 hd][64 keys]
      const int r = tid >> 2, c0 = (tid & 3) * 16;
      async_copy32(&Ks[r * 72 + c0],
                   k + ((long)(b * S_ + kt + r)) * (KV_ * HD_) + kvh * HD_ + c0);
      async_copy32(&Vs[r * 72 + c0],
                   vT + (((long)(b * KV_ + kvh) * HD_ + r) * S_) + kt + c0);
    }
    async_wait();
    __syncthreads();

    // scores: 4 tiles of 16 keys, K-dim = HD (2 WMMA steps)
    v8f sc[4];
#pragma unroll
    for (int nj = 0; nj < 4; ++nj){
#pragma unroll
      for (int r = 0; r < 8; ++r) sc[nj][r] = 0.f;
#pragma unroll
      for (int f = 0; f < 2; ++f){
        v16bf bf = fragB(&Ks[(nj * 16 + (lane & 15)) * 72 + f * 32 + kb]);
        sc[nj] = wmma_bf16(qf[f], bf, sc[nj]);
      }
    }
    // online softmax (row stats across 16 lanes of each half-wave)
#pragma unroll
    for (int r = 0; r < 8; ++r){
      const int qRow = qBase + mb + r;
      float mx = -1e30f;
#pragma unroll
      for (int nj = 0; nj < 4; ++nj){
        float sv = sc[nj][r] * 0.125f;                 // 1/sqrt(HD)
        if (kt + nj * 16 + (lane & 15) > qRow) sv = -1e9f;
        sc[nj][r] = sv;
        mx = fmaxf(mx, sv);
      }
#pragma unroll
      for (int msk = 1; msk < 16; msk <<= 1) mx = fmaxf(mx, __shfl_xor(mx, msk, 32));
      const float mnew = fmaxf(mrow[r], mx);
      const float corr = __expf(mrow[r] - mnew);
      mrow[r] = mnew;
      float rs = 0.f;
#pragma unroll
      for (int nj = 0; nj < 4; ++nj){
        const float pv = __expf(sc[nj][r] - mnew);
        rs += pv;
        Ps[(w * 16 + mb + r) * 72 + nj * 16 + (lane & 15)] = f2bf(pv);
      }
#pragma unroll
      for (int msk = 1; msk < 16; msk <<= 1) rs += __shfl_xor(rs, msk, 32);
      lrow[r] = lrow[r] * corr + rs;
#pragma unroll
      for (int t4 = 0; t4 < 4; ++t4) oa[t4][r] *= corr;
    }
    // O += P @ V  (P re-layouted via wave-private LDS into A-frags)
    v16bf pf[2];
    pf[0] = fragA(&Ps[(w * 16 + (lane & 15)) * 72 + kl]);
    pf[1] = fragA(&Ps[(w * 16 + (lane & 15)) * 72 + 32 + kl]);
#pragma unroll
    for (int t4 = 0; t4 < 4; ++t4)
#pragma unroll
      for (int f = 0; f < 2; ++f){
        v16bf bf = fragB(&Vs[(t4 * 16 + (lane & 15)) * 72 + f * 32 + kb]);
        oa[t4] = wmma_bf16(pf[f], bf, oa[t4]);
      }
    __syncthreads();
  }

#pragma unroll
  for (int r = 0; r < 8; ++r){
    const float inv = 1.0f / lrow[r];
#pragma unroll
    for (int t4 = 0; t4 < 4; ++t4)
      o[((long)(b * S_ + qBase + mb + r)) * (H_ * HD_) + hh * HD_ + t4 * 16 + (lane & 15)]
          = f2bf(oa[t4][r] * inv);
  }
}

__global__ void zero_kernel(int* c){ if (threadIdx.x < E_) c[threadIdx.x] = 0; }

// =====================================================================
// Host launch
// =====================================================================
extern "C" void kernel_launch(void* const* d_in, const int* in_sizes, int n_in,
                              void* d_out, int out_size, void* d_ws, size_t ws_size,
                              hipStream_t stream){
  (void)in_sizes; (void)n_in; (void)out_size; (void)ws_size;
  const float* x      = (const float*)d_in[0];
  const float* cosb   = (const float*)d_in[1];
  const float* sinb   = (const float*)d_in[2];
  /* d_in[3] = mask : causal, computed analytically */
  const float* rw_a   = (const float*)d_in[4];
  const float* rw_m   = (const float*)d_in[5];
  const float* Wq     = (const float*)d_in[6];
  const float* Wk     = (const float*)d_in[7];
  const float* Wv     = (const float*)d_in[8];
  const float* Wo     = (const float*)d_in[9];
  const float* router = (const float*)d_in[10];
  const float* Wg     = (const float*)d_in[11];
  const float* Wu     = (const float*)d_in[12];
  const float* Wd     = (const float*)d_in[13];

  float* outO = (float*)d_out;
  float* outK = outO + (long)NTOK * D_;
  float* outV = outK + (long)NTOK * KV_ * HD_;

  char* ws = (char*)d_ws;
  size_t off = 0;
  auto alloc = [&](size_t bytes) -> void* {
    void* p = ws + off; off += (bytes + 255) & ~(size_t)255; return p;
  };
  __bf16* xb  = (__bf16*)alloc((size_t)NTOK * D_ * 2);            // reused: attn out
  __bf16* qb  = (__bf16*)alloc((size_t)NTOK * D_ * 2);            // reused: y
  __bf16* kb  = (__bf16*)alloc((size_t)NTOK * KV_ * HD_ * 2);
  __bf16* vT  = (__bf16*)alloc((size_t)B_ * KV_ * HD_ * S_ * 2);
  float*  hb  = (float*) alloc((size_t)NTOK * D_ * 4);
  __bf16* he  = (__bf16*)alloc((size_t)E_ * CAP * F_ * 2);
  int*    cnt = (int*)   alloc(E_ * 4);
  int*    lst = (int*)   alloc((size_t)E_ * CAP * 4);
  float*  gl  = (float*) alloc((size_t)E_ * CAP * 4);
  __bf16* attn = xb;   // reuse after QKV projections consume xb
  __bf16* yb   = qb;   // reuse after attention consumes qb

  zero_kernel<<<1, 32, 0, stream>>>(cnt);

  // 1) RMSNorm -> bf16
  rmsnorm_kernel<<<NTOK, 256, 0, stream>>>(x, rw_a, xb);

  // 2) Q/K/V projections (WMMA GEMMs)
  GemmP pq = {}; pq.mode = 0; pq.A = xb; pq.lda = D_; pq.W = Wq; pq.ldw = H_*HD_;
  pq.K = D_; pq.outB = qb; pq.ldo = H_*HD_;
  gemm_kernel<<<dim3(NTOK/BM, (H_*HD_)/BN, 1), 256, 0, stream>>>(pq);

  GemmP pk = {}; pk.mode = 0; pk.A = xb; pk.lda = D_; pk.W = Wk; pk.ldw = KV_*HD_;
  pk.K = D_; pk.outB = kb; pk.ldo = KV_*HD_;
  gemm_kernel<<<dim3(NTOK/BM, (KV_*HD_)/BN, 1), 256, 0, stream>>>(pk);

  GemmP pv = {}; pv.mode = 1; pv.A = xb; pv.lda = D_; pv.W = Wv; pv.ldw = KV_*HD_;
  pv.K = D_; pv.vF = outV; pv.vT = vT;
  gemm_kernel<<<dim3(NTOK/BM, (KV_*HD_)/BN, 1), 256, 0, stream>>>(pv);

  // 3) RoPE (q in-place; k in-place + f32 k output)
  rope_q_kernel<<<(NTOK*H_*32)/256, 256, 0, stream>>>(qb, cosb, sinb);
  rope_k_kernel<<<(NTOK*KV_*32)/256, 256, 0, stream>>>(kb, cosb, sinb, outK);

  // 4) causal GQA flash attention
  attn_kernel<<<dim3(S_/128, H_, B_), 256, 0, stream>>>(qb, kb, vT, attn);

  // 5) output projection + residual -> h (ws) and d_out (MoE accumulates onto it)
  GemmP po = {}; po.mode = 2; po.A = attn; po.lda = D_; po.W = Wo; po.ldw = D_;
  po.K = D_; po.resid = x; po.outF = hb; po.outF2 = outO;
  gemm_kernel<<<dim3(NTOK/BM, D_/BN, 1), 256, 0, stream>>>(po);

  // 6) RMSNorm + router top-2 + expert token lists
  rms_router_kernel<<<NTOK, 256, 0, stream>>>(hb, rw_m, router, yb, cnt, lst, gl);

  // 7) MoE stage1: he = silu(y@Wg)*(y@Wu) per expert (gathered rows, dual GEMM)
  GemmP p1 = {}; p1.mode = 3; p1.A = yb; p1.lda = D_; p1.W = Wg; p1.W2 = Wu;
  p1.w_estride = (long)D_ * F_; p1.ldw = F_; p1.K = D_;
  p1.outB = he; p1.ldo = F_; p1.rowlist = lst; p1.cnts = cnt;
  gemm_kernel<<<dim3(CAP/BM, F_/BN, E_), 256, 0, stream>>>(p1);

  // 8) MoE stage2: out += gate * (he @ Wd), atomic scatter per token
  GemmP p2 = {}; p2.mode = 4; p2.A = he; p2.lda = F_; p2.a_estride = (long)CAP * F_;
  p2.W = Wd; p2.w_estride = (long)F_ * D_; p2.ldw = D_; p2.K = F_;
  p2.rowlist = lst; p2.glist = gl; p2.cnts = cnt; p2.scat = outO;
  gemm_kernel<<<dim3(CAP/BM, D_/BN, E_), 256, 0, stream>>>(p2);
}
